// SlayerLoihiMLP_9096740733669
// MI455X (gfx1250) — matrix operand order repaired and checked by reading, not compiled, not run on hardware
//
#include <hip/hip_runtime.h>
#include <hip/hip_bf16.h>

typedef __attribute__((ext_vector_type(16))) _Float16 v16h;
typedef __attribute__((ext_vector_type(8)))  float    v8f;

// ---------------------------------------------------------------------------
// f32 -> f16 converter (vectorized x4), same layout
// ---------------------------------------------------------------------------
__global__ __launch_bounds__(256) void cvt_f32_f16(const float* __restrict__ src,
                                                   _Float16* __restrict__ dst, int n) {
  int i = (blockIdx.x * 256 + threadIdx.x) * 4;
  if (i < n) {
    float4 f = *(const float4*)(src + i);
    dst[i + 0] = (_Float16)f.x;
    dst[i + 1] = (_Float16)f.y;
    dst[i + 2] = (_Float16)f.z;
    dst[i + 3] = (_Float16)f.w;
  }
}

// ---------------------------------------------------------------------------
// Fused convert + transpose: X f32 [b][CI][T] -> Y f16 [b][T][CI].
// 32x32 tiles through LDS (33-wide to avoid bank conflicts).
// Puts the layer-1 B operand directly into the [n][k] order the GEMM wants,
// so the GEMM never does a scalar transpose (and never redundantly per m-block).
// ---------------------------------------------------------------------------
__global__ __launch_bounds__(256) void cvt_transpose_f16(const float* __restrict__ X,
                                                         _Float16* __restrict__ Y,
                                                         int CI, int T) {
  __shared__ _Float16 tile[32][33];
  const int t0 = blockIdx.x * 32;
  const int i0 = blockIdx.y * 32;
  const int b  = blockIdx.z;
  const float* Xp = X + (size_t)b * CI * T;
  _Float16*    Yp = Y + (size_t)b * CI * T;

  const int lt = threadIdx.x & 31;   // t within tile (coalesced read dim)
  const int lr = threadIdx.x >> 5;   // row group
  #pragma unroll
  for (int r = 0; r < 32; r += 8)
    tile[lr + r][lt] = (_Float16)Xp[(size_t)(i0 + lr + r) * T + (t0 + lt)];
  __syncthreads();
  const int wi = threadIdx.x & 31;   // i within tile (coalesced write dim)
  const int wt = threadIdx.x >> 5;
  #pragma unroll
  for (int r = 0; r < 32; r += 8)
    Yp[(size_t)(t0 + wt + r) * CI + (i0 + wi)] = tile[wi][wt + r];
}

// ---------------------------------------------------------------------------
// Batched WMMA GEMM:  C[b][n][m] = sum_k A[m][k] * B[b][n][k]
// A: weights [M][K] f16 (shared over batch). B: [b][N][K] f16. C: [b][N][M] f32.
// Block: 256 threads = 8 waves; tile BM=64 x BN=128 x BK=32.
// Each wave computes a 32x32 patch = 2x2 v_wmma_f32_16x16x32_f16 tiles.
// All staging is vectorized 16B loads; per-thread pointers hoisted out of the
// K-loop (advance by BK) so the inner loop has no 64-bit address multiplies.
// ---------------------------------------------------------------------------
#define GEMM_BM 64
#define GEMM_BN 128
#define GEMM_BK 32

__global__ __launch_bounds__(256) void gemm_wmma_kernel(
    const _Float16* __restrict__ A,
    const _Float16* __restrict__ B,
    float* __restrict__ C,
    int M, int N, int K) {
  __shared__ _Float16 As[GEMM_BM][GEMM_BK];   // 4 KB
  __shared__ _Float16 Bs[GEMM_BN][GEMM_BK];   // 8 KB, [n][k]

  const int tid = threadIdx.x;
  const int m0  = blockIdx.x * GEMM_BM;
  const int n0  = blockIdx.y * GEMM_BN;
  const int b   = blockIdx.z;
  float* Cp = C + (size_t)b * N * M;

  // ---- hoisted staging pointers ----
  // A tile: row = tid>>2 (0..63), col = (tid&3)*8 -> one uint4 per thread
  const _Float16* pA = A + (size_t)(m0 + (tid >> 2)) * K + ((tid & 3) * 8);
  _Float16* sA = &As[tid >> 2][(tid & 3) * 8];
  // B tile: row n = tid>>1 (0..127), col = (tid&1)*16 -> two uint4 per thread
  const _Float16* pB = B + (size_t)b * K * N +
                       (size_t)(n0 + (tid >> 1)) * K + ((tid & 1) * 16);
  _Float16* sB = &Bs[tid >> 1][(tid & 1) * 16];

  const int wid = tid >> 5, lane = tid & 31;
  const int wm = wid & 1, wn = wid >> 1;       // 2x4 wave grid
  const int lrow = lane & 15, hi = lane >> 4;  // lane row / lane half

  v8f acc[2][2] = {};

  for (int k0 = 0; k0 < K; k0 += GEMM_BK) {
    *(uint4*)sA       = *(const uint4*)pA;
    *(uint4*)sB       = *(const uint4*)pB;
    *(uint4*)(sB + 8) = *(const uint4*)(pB + 8);
    if (k0 + GEMM_BK < K) {                    // global_prefetch next K-chunk
      __builtin_prefetch(pA + GEMM_BK, 0, 1);
      __builtin_prefetch(pB + GEMM_BK, 0, 1);
    }
    pA += GEMM_BK;
    pB += GEMM_BK;
    __syncthreads();

    // ---- fragments from LDS (CDNA5 16-bit WMMA register layouts) ----
    v16h afr[2], bfr[2];
    #pragma unroll
    for (int mi = 0; mi < 2; ++mi) {
      int m  = wm * 32 + mi * 16 + lrow;
      int kb = hi * 8;                          // lane half: K+8 / K+24 groups
      union { uint32_t u[8]; v16h v; } tmp;
      #pragma unroll
      for (int p = 0; p < 8; ++p) {
        int kk = kb + ((p < 4) ? (2 * p) : (16 + 2 * (p - 4)));
        tmp.u[p] = *(const uint32_t*)&As[m][kk];  // merges to 2x ds_load_b128
      }
      afr[mi] = tmp.v;
    }
    #pragma unroll
    for (int ni = 0; ni < 2; ++ni) {
      int n  = wn * 32 + ni * 16 + lrow;
      int kb = hi * 8;                          // lane half: K+8 / K+24 groups
      union { uint4 q[2]; v16h v; } tmp;
      tmp.q[0] = *(const uint4*)&Bs[n][kb];
      tmp.q[1] = *(const uint4*)&Bs[n][kb + 16];
      bfr[ni] = tmp.v;
    }

    #pragma unroll
    for (int mi = 0; mi < 2; ++mi)
      #pragma unroll
      for (int ni = 0; ni < 2; ++ni)
        acc[mi][ni] = __builtin_amdgcn_wmma_f32_16x16x32_f16(
            false, afr[mi], false, bfr[ni], (short)0, acc[mi][ni], false, false);
    __syncthreads();
  }

  // ---- store: lane's 8 accs are 8 consecutive M at fixed N -> two float4 ----
  #pragma unroll
  for (int mi = 0; mi < 2; ++mi)
    #pragma unroll
    for (int ni = 0; ni < 2; ++ni) {
      int n = n0 + wn * 32 + ni * 16 + lrow;
      int m = m0 + wm * 32 + mi * 16 + 8 * hi;
      union { v8f v; float4 f[2]; } t;
      t.v = acc[mi][ni];
      float* dst = Cp + (size_t)n * M + m;
      *(float4*)dst       = t.f[0];
      *(float4*)(dst + 4) = t.f[1];
    }
}

// ---------------------------------------------------------------------------
// psp: causal FIR along T with exact Loihi SRM taps (generated on device with
// the reference's integer recurrence; K ~= 88, bounded by KMAX).
// X,Y layout [b][t][c] (c contiguous). Block tiles 32 channels x 128 t-steps.
// ---------------------------------------------------------------------------
#define KMAX 128
#define FTB  128
#define FOB  32

__global__ __launch_bounds__(256) void psp_fir_kernel(const float* __restrict__ X,
                                                      float* __restrict__ Y,
                                                      int C, int T) {
  __shared__ float ker[KMAX];
  __shared__ int   Kn_s;
  __shared__ float xs[FTB + KMAX][FOB];   // 32 KB

  const int tid = threadIdx.x;
  if (tid == 0) {
    // ker = [64]; while ker[-1] > 0: u = u*3072>>12; v = (v*3968>>12)+u; append
    int u = 64, v = 64, k = 0;
    ker[k++] = 1.0f;                      // 64/64
    while (v > 0 && k < KMAX) {
      u = (u * 3072) >> 12;
      v = ((v * 3968) >> 12) + u;
      ker[k++] = (float)v * (1.0f / 64.0f);
    }
    Kn_s = k;
    for (int i = k; i < KMAX; ++i) ker[i] = 0.0f;
  }

  const int o0 = blockIdx.x * FOB;
  const int t0 = blockIdx.y * FTB;
  const int b  = blockIdx.z;
  const float* Xp = X + (size_t)b * T * C;
  float*       Yp = Y + (size_t)b * T * C;

  // load tile + causal halo (zero-padded on the left)
  for (int idx = tid; idx < (FTB + KMAX) * FOB; idx += 256) {
    int tt = idx >> 5, oo = idx & 31;
    int t  = t0 - KMAX + tt;
    xs[tt][oo] = (t >= 0) ? Xp[(size_t)t * C + (o0 + oo)] : 0.0f;
  }
  __syncthreads();
  const int Kn = Kn_s;

  const int oo = tid & 31;
  const int tb = tid >> 5;                // 0..7
  for (int i = 0; i < FTB / 8; ++i) {
    int tt = i * 8 + tb;
    float acc = 0.0f;
    for (int k = 0; k < Kn; ++k)
      acc += ker[k] * xs[KMAX + tt - k][oo];   // conflict-free: lanes span banks
    Yp[(size_t)(t0 + tt) * C + (o0 + oo)] = acc;
  }
}

// ---------------------------------------------------------------------------
// Loihi LIF scan (exact integer dynamics) + delay_shift.
// One thread per (b, channel); lockstep t => coalesced reads of [b][t][c].
// ---------------------------------------------------------------------------
__device__ __forceinline__ int lif_decay(int x, int mul) {
  int s = (x >= 0) ? 1 : -1;
  return s * ((s * x * mul) >> 12);
}

// layer-1 variant: spikes out as f16 in [b][t][c] (B-operand of GEMM2)
__global__ __launch_bounds__(256) void lif_scan_f16(const float* __restrict__ Y,
                                                    _Float16* __restrict__ S,
                                                    int C, int T) {
  const int o = blockIdx.x * 256 + threadIdx.x;
  const int b = blockIdx.y;
  const float* Yp = Y + (size_t)b * T * C + o;
  _Float16*    Sp = S + (size_t)b * T * C + o;
  Sp[0] = (_Float16)0.0f;                       // delay_shift: s[0] = 0
  int u = 0, v = 0;
  for (int t = 0; t < T; ++t) {
    int xi = (int)truncf(Yp[(size_t)t * C] * 64.0f);   // trunc(w * WGT_SCALE)
    u = lif_decay(u, 3072) + xi;                        // 4096 - I_DECAY
    v = lif_decay(v, 3968) + u;                         // 4096 - V_DECAY
    int sp = (v >= 5120);                               // THETA = 80<<6
    if (sp) v = 0;
    if (t + 1 < T) Sp[(size_t)(t + 1) * C] = (_Float16)(float)sp;
  }
}

// layer-2 variant: spikes out as f32 in [b][c][t] (final output layout)
__global__ __launch_bounds__(256) void lif_scan_out(const float* __restrict__ Y,
                                                    float* __restrict__ O,
                                                    int C, int T) {
  const int o = blockIdx.x * 256 + threadIdx.x;
  const int b = blockIdx.y;
  const float* Yp = Y + (size_t)b * T * C + o;
  float*       Op = O + ((size_t)b * C + o) * T;
  Op[0] = 0.0f;
  int u = 0, v = 0;
  for (int t = 0; t < T; ++t) {
    int xi = (int)truncf(Yp[(size_t)t * C] * 64.0f);
    u = lif_decay(u, 3072) + xi;
    v = lif_decay(v, 3968) + u;
    int sp = (v >= 5120);
    if (sp) v = 0;
    if (t + 1 < T) Op[t + 1] = (float)sp;
  }
}

// ---------------------------------------------------------------------------
extern "C" void kernel_launch(void* const* d_in, const int* in_sizes, int n_in,
                              void* d_out, int out_size, void* d_ws, size_t ws_size,
                              hipStream_t stream) {
  constexpr int Bb = 32, CIN = 512, HID = 1024, COUT = 256, T = 512;

  const float* x  = (const float*)d_in[0];   // [b][cin][t] spikes (0/1)
  const float* W1 = (const float*)d_in[1];   // [hid][cin]
  const float* W2 = (const float*)d_in[2];   // [cout][hid]

  char* ws = (char*)d_ws;
  size_t off = 0;
  _Float16* W1h = (_Float16*)(ws + off);  off += (size_t)HID * CIN * 2;       // 1 MB
  _Float16* W2h = (_Float16*)(ws + off);  off += (size_t)COUT * HID * 2;      // 0.5 MB
  size_t xh_off = off;
  _Float16* xhT = (_Float16*)(ws + off);  off += (size_t)Bb * CIN * T * 2;    // 16.8 MB, [b][t][i]
  size_t z1_off = off;
  float* Z1 = (float*)(ws + off);         off += (size_t)Bb * T * HID * 4;    // 67 MB
  float* Y1 = (float*)(ws + off);         off += (size_t)Bb * T * HID * 4;    // 67 MB
  _Float16* S1 = (_Float16*)(ws + off);   off += (size_t)Bb * T * HID * 2;    // 33.5 MB
  // reuse dead regions: Z2 aliases xhT (exactly 16.8 MB), Y2 aliases Z1
  float* Z2 = (float*)(ws + xh_off);
  float* Y2 = (float*)(ws + z1_off);
  float* out = (float*)d_out;                                                  // [b][cout][t]

  // 1) precision staging: weights f32->f16; spikes f32->f16 WITH transpose to
  //    [b][t][i] so both GEMMs consume B as [n][k] with 16B vector loads.
  cvt_f32_f16<<<(HID * CIN) / 1024, 256, 0, stream>>>(W1, W1h, HID * CIN);
  cvt_f32_f16<<<(COUT * HID) / 1024, 256, 0, stream>>>(W2, W2h, COUT * HID);
  cvt_transpose_f16<<<dim3(T / 32, CIN / 32, Bb), 256, 0, stream>>>(x, xhT, CIN, T);

  // 2) GEMM1: Z1[b][t][o] = sum_i W1[o][i] * x[b][i][t]
  gemm_wmma_kernel<<<dim3(HID / GEMM_BM, T / GEMM_BN, Bb), 256, 0, stream>>>(
      W1h, xhT, Z1, HID, T, CIN);

  // 3) psp FIR + LIF scan + delay (layer 1) -> f16 spikes [b][t][h]
  psp_fir_kernel<<<dim3(HID / FOB, T / FTB, Bb), 256, 0, stream>>>(Z1, Y1, HID, T);
  lif_scan_f16<<<dim3(HID / 256, Bb), 256, 0, stream>>>(Y1, S1, HID, T);

  // 4) GEMM2: Z2[b][t][o2] = sum_h W2[o2][h] * s1[b][t][h]
  gemm_wmma_kernel<<<dim3(COUT / GEMM_BM, T / GEMM_BN, Bb), 256, 0, stream>>>(
      W2h, S1, Z2, COUT, T, HID);

  // 5) psp FIR + LIF scan + delay (layer 2) -> final f32 output [b][c][t]
  psp_fir_kernel<<<dim3(COUT / FOB, T / FTB, Bb), 256, 0, stream>>>(Z2, Y2, COUT, T);
  lif_scan_out<<<dim3(COUT / 256, Bb), 256, 0, stream>>>(Y2, out, COUT, T);
}
